// IPFECNN_67284957659694
// MI455X (gfx1250) — matrix operand order repaired and checked by reading, not compiled, not run on hardware
//
#include <hip/hip_runtime.h>
#include <hip/hip_bf16.h>

typedef __attribute__((ext_vector_type(16))) _Float16 v16h;
typedef __attribute__((ext_vector_type(8)))  _Float16 v8h;
typedef __attribute__((ext_vector_type(8)))  float    v8f;

#define BN_EPS 1e-5f
#define QSCALE 10000.0f
#define PMOD   1000000006.0f   // P - 1

// ---- CDNA5 async global->LDS copy (ASYNCcnt) with safe fallback ------------
// Probe round 2 showed the builtin exists with param0 type:
//   "__attribute__((vector_size(4*sizeof(int)))) int __device__ *"  (int4 AS1*)
#if defined(__AMDGCN__) && __has_builtin(__builtin_amdgcn_global_load_async_to_lds_b128)
#define HAS_ASYNC_LDS 1
typedef int async_v4i __attribute__((vector_size(16)));
#define COPY16(gp, lp)                                                         \
    __builtin_amdgcn_global_load_async_to_lds_b128(                            \
        (__attribute__((address_space(1))) async_v4i*)(_Float16*)(gp),         \
        (__attribute__((address_space(3))) async_v4i*)(lp), 0, 0)
#if __has_builtin(__builtin_amdgcn_s_wait_asynccnt)
#define WAIT_COPIES() __builtin_amdgcn_s_wait_asynccnt(0)
#else
#define WAIT_COPIES() asm volatile("s_wait_asynccnt 0x0" ::: "memory")
#endif
#else
#define HAS_ASYNC_LDS 0
#define COPY16(gp, lp) (*(uint4*)(lp) = *(const uint4*)(gp))
#define WAIT_COPIES() ((void)0)
#endif

// ---------------------------------------------------------------------------
// Fold bias + batchnorm into per-channel scale/shift:
//   y = acc * (inv * qinv) + (bias*inv + beta - mean*inv),  inv = g*rsqrt(v+eps)
// ---------------------------------------------------------------------------
__global__ void affine_fold_kernel(const float* __restrict__ g, const float* __restrict__ beta,
                                   const float* __restrict__ mean, const float* __restrict__ var,
                                   const float* __restrict__ bias, float qinv,
                                   float* __restrict__ scale, float* __restrict__ shift, int n)
{
    int i = blockIdx.x * blockDim.x + threadIdx.x;
    if (i >= n) return;
    float inv = g[i] * rsqrtf(var[i] + BN_EPS);
    scale[i] = inv * qinv;
    shift[i] = bias[i] * inv + (beta[i] - mean[i] * inv);
}

__global__ void affine_ident_kernel(const float* __restrict__ bias,
                                    float* __restrict__ scale, float* __restrict__ shift, int n)
{
    int i = blockIdx.x * blockDim.x + threadIdx.x;
    if (i >= n) return;
    scale[i] = 1.0f;
    shift[i] = bias[i];
}

// ---------------------------------------------------------------------------
// Pack OIHW fp32 conv weights -> fp16 [N][K] with K = (ky*3+kx)*C + c
// ---------------------------------------------------------------------------
__global__ void pack_conv_w_kernel(const float* __restrict__ w, _Float16* __restrict__ wt,
                                   int N, int C)
{
    int idx = blockIdx.x * blockDim.x + threadIdx.x;
    int total = N * C * 9;
    if (idx >= total) return;
    int kx = idx % 3; int t = idx / 3;
    int ky = t % 3;   t /= 3;
    int c  = t % C;   int n = t / C;
    wt[(size_t)n * (9 * C) + (size_t)(ky * 3 + kx) * C + c] = (_Float16)w[idx];
}

// fc1_w [128,2304], src k = c*9 + y*3 + x (NCHW flatten) -> NHWC k' = (y*3+x)*256 + c
__global__ void pack_fc1_w_kernel(const float* __restrict__ w, _Float16* __restrict__ wt)
{
    int idx = blockIdx.x * blockDim.x + threadIdx.x;
    int total = 128 * 2304;
    if (idx >= total) return;
    int k = idx % 2304; int n = idx / 2304;
    int xp = k % 3; int t = k / 3;
    int yp = t % 3; int c = t / 3;
    wt[(size_t)n * 2304 + (size_t)(yp * 3 + xp) * 256 + c] = (_Float16)w[idx];
}

// ---------------------------------------------------------------------------
// conv1 (Cin=1) fused: IPFE dequant + conv + BN + ReLU + 2x2 maxpool.
// Quantized weights / affine staged in LDS once per block.
// ---------------------------------------------------------------------------
__global__ void conv1_fused_kernel(const float* __restrict__ x, const float* __restrict__ w,
                                   const float* __restrict__ scale, const float* __restrict__ shift,
                                   _Float16* __restrict__ out, int Bv)
{
    __shared__ float swq[64 * 9];
    __shared__ float ssc[64];
    __shared__ float ssh[64];
    int tid = threadIdx.x;
    if (tid < 64) { ssc[tid] = scale[tid]; ssh[tid] = shift[tid]; }
    for (int j = tid; j < 576; j += 256) swq[j] = roundf(w[j] * QSCALE);
    __syncthreads();

    int idx = blockIdx.x * blockDim.x + tid;
    int total = Bv * 14 * 14;
    if (idx >= total) return;
    int xo = idx % 14; int t = idx / 14;
    int yo = t % 14;   int b = t / 14;

    float patch[4][4];
    int y0 = 2 * yo - 1, x0 = 2 * xo - 1;
    #pragma unroll
    for (int i = 0; i < 4; ++i)
        #pragma unroll
        for (int j = 0; j < 4; ++j) {
            int yy = y0 + i, xx = x0 + j;
            float val = 0.0f;
            if (yy >= 0 && yy < 28 && xx >= 0 && xx < 28) {
                float raw = x[((size_t)b * 28 + yy) * 28 + xx];
                float tq = truncf(raw);
                tq = fmodf(tq, PMOD);
                if (tq < 0.0f) tq += PMOD;
                val = tq;
            }
            patch[i][j] = val;
        }

    _Float16* dst = out + (size_t)idx * 64;
    for (int n = 0; n < 64; ++n) {
        const float* wq = &swq[n * 9];
        float s00 = 0.f, s01 = 0.f, s10 = 0.f, s11 = 0.f;
        #pragma unroll
        for (int ky = 0; ky < 3; ++ky)
            #pragma unroll
            for (int kx = 0; kx < 3; ++kx) {
                float wv = wq[ky * 3 + kx];
                s00 += patch[ky][kx]         * wv;
                s01 += patch[ky][kx + 1]     * wv;
                s10 += patch[ky + 1][kx]     * wv;
                s11 += patch[ky + 1][kx + 1] * wv;
            }
        float sc = ssc[n], sh = ssh[n];
        float v0 = fmaxf(s00 * sc + sh, 0.f);
        float v1 = fmaxf(s01 * sc + sh, 0.f);
        float v2 = fmaxf(s10 * sc + sh, 0.f);
        float v3 = fmaxf(s11 * sc + sh, 0.f);
        dst[n] = (_Float16)fmaxf(fmaxf(v0, v1), fmaxf(v2, v3));
    }
}

// ---------------------------------------------------------------------------
// Implicit-GEMM conv via WMMA f16->f32, double-buffered LDS, async global->LDS.
//   A[m,k] = im2col(src NHWC),  B stored transposed as wt[n][k].
// Block: 256 threads / 8 waves, 128x128 tile; wave tile 32x64 (2x4 v8f accs).
// Requires M%128==0, N%128==0, K%32==0, Cin%32==0 (all hold here).
// Steady-state loop: no integer division; tap (ky,kx,c0) walked incrementally.
// ---------------------------------------------------------------------------
__launch_bounds__(256)
__global__ void wmma_gemm_conv_kernel(const _Float16* __restrict__ src,
                                      const _Float16* __restrict__ wt,
                                      const float* __restrict__ scale,
                                      const float* __restrict__ shift,
                                      _Float16* __restrict__ dst,
                                      int Hin, int Win, int Cin, int ksize,
                                      int Nout, int Ktot, int relu)
{
    __shared__ _Float16 lA[2][128 * 32];   // 2 x 8 KB
    __shared__ _Float16 lB[2][128 * 32];   // 2 x 8 KB

    const int tid     = threadIdx.x;
    const int lane    = tid & 31;
    const int wv      = tid >> 5;
    const int waveM   = wv & 3;
    const int waveN   = wv >> 2;
    const int laneRow = lane & 15;
    const int halfSel = lane >> 4;

    const long long mBase = (long long)blockIdx.x * 128;
    const int nBase = blockIdx.y * 128;
    const int pad = ksize >> 1;
    const int HW = Hin * Win;

    // ---- hoisted staging-thread geometry (divisions done once) ----
    const int row0  = tid >> 2;            // 0..63
    const int row1  = row0 + 64;           // 64..127
    const int chunk = (tid & 3) * 8;       // half offset of 16B chunk
    long long m0 = mBase + row0;
    long long m1 = mBase + row1;
    const int b0 = (int)(m0 / HW); int r0 = (int)(m0 - (long long)b0 * HW);
    const int b1 = (int)(m1 / HW); int r1 = (int)(m1 - (long long)b1 * HW);
    const int yA0 = r0 / Win, xA0 = r0 % Win;
    const int yA1 = r1 / Win, xA1 = r1 % Win;
    const size_t wRow0 = (size_t)(nBase + row0) * Ktot + chunk;
    const size_t wRow1 = (size_t)(nBase + row1) * Ktot + chunk;
    uint4 zero4; zero4.x = zero4.y = zero4.z = zero4.w = 0u;

    // stage one 32-wide K slice (tap dy,dx; channel base c0) into buffer `buf`
    auto stage = [&](int buf, int dy, int dx, int c0, int kOff) {
        {   // A row0
            int yy = yA0 + dy, xx = xA0 + dx;
            _Float16* lp = &lA[buf][row0 * 32 + chunk];
            if (yy >= 0 && yy < Hin && xx >= 0 && xx < Win)
                COPY16(src + (((size_t)b0 * Hin + yy) * Win + xx) * Cin + c0 + chunk, lp);
            else
                *(uint4*)lp = zero4;
        }
        {   // A row1
            int yy = yA1 + dy, xx = xA1 + dx;
            _Float16* lp = &lA[buf][row1 * 32 + chunk];
            if (yy >= 0 && yy < Hin && xx >= 0 && xx < Win)
                COPY16(src + (((size_t)b1 * Hin + yy) * Win + xx) * Cin + c0 + chunk, lp);
            else
                *(uint4*)lp = zero4;
        }
        // B: two contiguous 16B chunks along K
        COPY16(wt + wRow0 + kOff, &lB[buf][row0 * 32 + chunk]);
        COPY16(wt + wRow1 + kOff, &lB[buf][row1 * 32 + chunk]);
    };

    // incremental tap walk (no division in steady state)
    int dyS = -pad, dxS = -pad, c0S = 0, kOffS = 0;
    auto advance = [&]() {
        c0S += 32;
        if (c0S >= Cin) { c0S = 0; dxS += 1; if (dxS > pad) { dxS = -pad; dyS += 1; } }
        kOffS += 32;
    };

    v8f acc[2][4] = {};

    const int totalIter = Ktot >> 5;
    stage(0, dyS, dxS, c0S, kOffS);        // prologue: fill buffer 0
    advance();
    WAIT_COPIES();
    __syncthreads();

    for (int it = 0; it < totalIter; ++it) {
        const int cur = it & 1;
        if (it + 1 < totalIter) {           // issue next tile while computing
            stage(cur ^ 1, dyS, dxS, c0S, kOffS);
            advance();
        }

        // ---- fragments per ISA 16-bit WMMA layouts ----
        v16h afrag[2], bfrag[4];
        #pragma unroll
        for (int ms = 0; ms < 2; ++ms) {
            int r = waveM * 32 + ms * 16 + laneRow;
            union { v16h v; v8h h[2]; } u;
            u.h[0] = *(const v8h*)(&lA[cur][r * 32 + halfSel * 8]);       // K 0..7 / 8..15
            u.h[1] = *(const v8h*)(&lA[cur][r * 32 + 16 + halfSel * 8]);  // K 16..23 / 24..31
            afrag[ms] = u.v;
        }
        #pragma unroll
        for (int ns = 0; ns < 4; ++ns) {
            int nn = waveN * 64 + ns * 16 + laneRow;
            union { v16h v; v8h h[2]; } u;
            u.h[0] = *(const v8h*)(&lB[cur][nn * 32 + halfSel * 8]);
            u.h[1] = *(const v8h*)(&lB[cur][nn * 32 + 16 + halfSel * 8]);
            bfrag[ns] = u.v;
        }

        #pragma unroll
        for (int ms = 0; ms < 2; ++ms)
            #pragma unroll
            for (int ns = 0; ns < 4; ++ns)
                acc[ms][ns] = __builtin_amdgcn_wmma_f32_16x16x32_f16(
                    false, afrag[ms], false, bfrag[ns],
                    (short)0, acc[ms][ns], false, false);

        WAIT_COPIES();        // my async copies into buf^1 landed
        __syncthreads();      // everyone's copies visible / reads of cur done
    }

    // ---- epilogue: scale/shift (+ReLU), fp16 store ----
    // C layout: VGPR r -> M = r + halfSel*8, N = lane%16
    #pragma unroll
    for (int ns = 0; ns < 4; ++ns) {
        int col = nBase + waveN * 64 + ns * 16 + laneRow;
        float sc = scale[col], sh = shift[col];
        #pragma unroll
        for (int ms = 0; ms < 2; ++ms) {
            long long rowBase = mBase + waveM * 32 + ms * 16 + halfSel * 8;
            #pragma unroll
            for (int r = 0; r < 8; ++r) {
                float vv = acc[ms][ns][r] * sc + sh;
                if (relu) vv = fmaxf(vv, 0.0f);
                dst[(size_t)(rowBase + r) * Nout + col] = (_Float16)vv;
            }
        }
    }
}

// ---------------------------------------------------------------------------
// 2x2/2 VALID maxpool on fp16 NHWC, vectorized over 8 channels (16B ld/st).
// ---------------------------------------------------------------------------
__global__ void maxpool2_kernel(const _Float16* __restrict__ src, _Float16* __restrict__ dst,
                                int Bv, int H, int W, int C)
{
    int Ho = H >> 1, Wo = W >> 1, C8 = C >> 3;
    long long idx = (long long)blockIdx.x * blockDim.x + threadIdx.x;
    long long total = (long long)Bv * Ho * Wo * C8;
    if (idx >= total) return;
    int c8 = (int)(idx % C8); long long t = idx / C8;
    int xo = (int)(t % Wo);   t /= Wo;
    int yo = (int)(t % Ho);   int b = (int)(t / Ho);
    const _Float16* p = src + (((size_t)b * H + 2 * yo) * W + 2 * xo) * C + c8 * 8;
    v8h a0 = *(const v8h*)(p);
    v8h a1 = *(const v8h*)(p + C);
    v8h a2 = *(const v8h*)(p + (size_t)W * C);
    v8h a3 = *(const v8h*)(p + (size_t)W * C + C);
    v8h r;
    #pragma unroll
    for (int i = 0; i < 8; ++i)
        r[i] = (_Float16)fmaxf(fmaxf((float)a0[i], (float)a1[i]),
                               fmaxf((float)a2[i], (float)a3[i]));
    *(v8h*)(dst + ((((size_t)b * Ho + yo) * Wo + xo) * C + c8 * 8)) = r;
}

// ---------------------------------------------------------------------------
// fc2: [B,128] fp16 x [10,128] fp32 + bias -> fp32 out.  N=10: VALU dot.
// ---------------------------------------------------------------------------
__global__ void fc2_kernel(const _Float16* __restrict__ a, const float* __restrict__ w,
                           const float* __restrict__ bias, float* __restrict__ out, int Bv)
{
    int idx = blockIdx.x * blockDim.x + threadIdx.x;
    int total = Bv * 10;
    if (idx >= total) return;
    int n = idx % 10, b = idx / 10;
    const _Float16* ap = a + (size_t)b * 128;
    const float* wp = w + n * 128;
    float s = 0.0f;
    #pragma unroll 8
    for (int k = 0; k < 128; ++k) s += (float)ap[k] * wp[k];
    out[idx] = s + bias[n];
}

// ---------------------------------------------------------------------------
extern "C" void kernel_launch(void* const* d_in, const int* in_sizes, int n_in,
                              void* d_out, int out_size, void* d_ws, size_t ws_size,
                              hipStream_t stream)
{
    const float* x       = (const float*)d_in[0];
    const float* conv1_w = (const float*)d_in[1];
    const float* conv1_b = (const float*)d_in[2];
    const float* bn1_g = (const float*)d_in[3],  *bn1_b = (const float*)d_in[4];
    const float* bn1_m = (const float*)d_in[5],  *bn1_v = (const float*)d_in[6];
    const float* conv2_w = (const float*)d_in[7];
    const float* conv2_b = (const float*)d_in[8];
    const float* bn2_g = (const float*)d_in[9],  *bn2_b = (const float*)d_in[10];
    const float* bn2_m = (const float*)d_in[11], *bn2_v = (const float*)d_in[12];
    const float* conv3_w = (const float*)d_in[13];
    const float* conv3_b = (const float*)d_in[14];
    const float* bn3_g = (const float*)d_in[15], *bn3_b = (const float*)d_in[16];
    const float* bn3_m = (const float*)d_in[17], *bn3_v = (const float*)d_in[18];
    const float* fc1_w = (const float*)d_in[19];
    const float* fc1_b = (const float*)d_in[20];
    const float* fc2_w = (const float*)d_in[21];
    const float* fc2_b = (const float*)d_in[22];

    const int B = in_sizes[0] / (28 * 28);      // 2048

    // ---- workspace layout (fp16 activations, reused regions) ----
    char* ws = (char*)d_ws;
    auto al = [](size_t v) { return (v + 255) & ~(size_t)255; };
    size_t n_p1 = (size_t)B * 14 * 14 * 64;     // == n_t3 (B*49*256)
    size_t n_t2 = (size_t)B * 14 * 14 * 128;
    size_t n_p2 = (size_t)B * 7 * 7 * 128;
    size_t n_p3 = (size_t)B * 3 * 3 * 256;

    size_t off = 0;
    size_t reg0 = off;  off = al(off + n_p1 * 2);   // p1, later reused as t3
    size_t reg1 = off;  off = al(off + n_t2 * 2);   // t2, later reused as {p3, t4}
    size_t regP2 = off; off = al(off + n_p2 * 2);
    size_t oW2 = off;   off = al(off + (size_t)128 * 576  * 2);
    size_t oW3 = off;   off = al(off + (size_t)256 * 1152 * 2);
    size_t oWF = off;   off = al(off + (size_t)128 * 2304 * 2);
    size_t oAff = off;

    _Float16* p1  = (_Float16*)(ws + reg0);
    _Float16* t3  = (_Float16*)(ws + reg0);
    _Float16* t2  = (_Float16*)(ws + reg1);
    _Float16* p3  = (_Float16*)(ws + reg1);
    _Float16* t4  = (_Float16*)(ws + reg1 + al(n_p3 * 2));
    _Float16* p2  = (_Float16*)(ws + regP2);
    _Float16* w2p = (_Float16*)(ws + oW2);
    _Float16* w3p = (_Float16*)(ws + oW3);
    _Float16* wfp = (_Float16*)(ws + oWF);
    float* sc1 = (float*)(ws + oAff);
    float* sh1 = sc1 + 64;
    float* sc2 = sh1 + 64;  float* sh2 = sc2 + 128;
    float* sc3 = sh2 + 128; float* sh3 = sc3 + 256;
    float* scf = sh3 + 256; float* shf = scf + 128;

    // ---- fold affine params ----
    affine_fold_kernel<<<1, 64, 0, stream>>>(bn1_g, bn1_b, bn1_m, bn1_v, conv1_b,
                                             1.0f / QSCALE, sc1, sh1, 64);
    affine_fold_kernel<<<1, 128, 0, stream>>>(bn2_g, bn2_b, bn2_m, bn2_v, conv2_b,
                                              1.0f, sc2, sh2, 128);
    affine_fold_kernel<<<1, 256, 0, stream>>>(bn3_g, bn3_b, bn3_m, bn3_v, conv3_b,
                                              1.0f, sc3, sh3, 256);
    affine_ident_kernel<<<1, 128, 0, stream>>>(fc1_b, scf, shf, 128);

    // ---- pack weights to fp16 [N][K] ----
    pack_conv_w_kernel<<<(128 * 64 * 9 + 255) / 256, 256, 0, stream>>>(conv2_w, w2p, 128, 64);
    pack_conv_w_kernel<<<(256 * 128 * 9 + 255) / 256, 256, 0, stream>>>(conv3_w, w3p, 256, 128);
    pack_fc1_w_kernel<<<(128 * 2304 + 255) / 256, 256, 0, stream>>>(fc1_w, wfp);

    // ---- conv1 + dequant + BN + ReLU + pool -> p1 [B,14,14,64] ----
    conv1_fused_kernel<<<(B * 196 + 255) / 256, 256, 0, stream>>>(x, conv1_w, sc1, sh1, p1, B);

    // ---- conv2: M = B*196, N = 128, K = 576 -> t2 [B,14,14,128] ----
    wmma_gemm_conv_kernel<<<dim3((unsigned)(B * 196 / 128), 1), 256, 0, stream>>>(
        p1, w2p, sc2, sh2, t2, 14, 14, 64, 3, 128, 576, 1);
    maxpool2_kernel<<<(unsigned)(((size_t)B * 7 * 7 * 16 + 255) / 256), 256, 0, stream>>>(
        t2, p2, B, 14, 14, 128);

    // ---- conv3: M = B*49, N = 256, K = 1152 -> t3 [B,7,7,256] ----
    wmma_gemm_conv_kernel<<<dim3((unsigned)(B * 49 / 128), 2), 256, 0, stream>>>(
        p2, w3p, sc3, sh3, t3, 7, 7, 128, 3, 256, 1152, 1);
    maxpool2_kernel<<<(unsigned)(((size_t)B * 3 * 3 * 32 + 255) / 256), 256, 0, stream>>>(
        t3, p3, B, 7, 7, 256);

    // ---- fc1 as 1x1 "conv": M = B, N = 128, K = 2304 -> t4 [B,128] ----
    wmma_gemm_conv_kernel<<<dim3((unsigned)(B / 128), 1), 256, 0, stream>>>(
        p3, wfp, scf, shf, t4, 1, 1, 2304, 1, 128, 2304, 1);

    // ---- fc2 -> d_out [B,10] fp32 ----
    fc2_kernel<<<(B * 10 + 255) / 256, 256, 0, stream>>>(t4, fc2_w, fc2_b, (float*)d_out, B);
}
